// Bilstm_CRF_MTL_31482110279728
// MI455X (gfx1250) — matrix operand order repaired and checked by reading, not compiled, not run on hardware
//
#include <hip/hip_runtime.h>
#include <hip/hip_bf16.h>

// ---------------------------------------------------------------- types
typedef __bf16 bf16;
typedef __attribute__((ext_vector_type(8)))  bf16  v8bf;
typedef __attribute__((ext_vector_type(16))) bf16  v16bf;
typedef __attribute__((ext_vector_type(8)))  float v8f;

#define B_  64
#define S_  256
#define V_  50000
#define E_  300
#define EP_ 320      // E padded to multiple of 32
#define H_  256
#define G4_ 1024     // 4*H
#define T_  22
#define STARTT 20
#define STOPT  21
#define ROWS_ (B_ * S_)   // 16384

__device__ __forceinline__ float sigf(float x) { return 1.f / (1.f + __expf(-x)); }

// A fragment: 16x32 bf16 tile, src points at (row0, k0) of row-major [*, ldk]
__device__ __forceinline__ v16bf load_frag_a(const bf16* src, int ldk, int lane) {
  int half = lane >> 4, m = lane & 15;
  const bf16* r = src + (size_t)m * ldk + (half ? 8 : 0);
  v8bf lo = *(const v8bf*)(r);
  v8bf hi = *(const v8bf*)(r + 16);
  return __builtin_shufflevector(lo, hi, 0,1,2,3,4,5,6,7,8,9,10,11,12,13,14,15);
}

// B fragment: 32x16 tile taken from W[N][K] row-major; src points at (n0, k0)
__device__ __forceinline__ v16bf load_frag_b(const bf16* src, int ldk, int lane) {
  int half = lane >> 4, n = lane & 15;
  const bf16* r = src + (size_t)n * ldk + half * 8;
  v8bf lo = *(const v8bf*)(r);
  v8bf hi = *(const v8bf*)(r + 16);
  return __builtin_shufflevector(lo, hi, 0,1,2,3,4,5,6,7,8,9,10,11,12,13,14,15);
}

__device__ __forceinline__ v8f wmma_bf16(v16bf a, v16bf b, v8f c) {
  return __builtin_amdgcn_wmma_f32_16x16x32_bf16(false, a, false, b, (short)0, c,
                                                 false, false);
}

// ---------------------------------------------------------------- K0: zero scalars
__global__ void k_zero(float* scal) {
  if (threadIdx.x < 4) scal[threadIdx.x] = 0.f;
}

// ---------------------------------------------------------------- K1: embedding gather -> bf16 padded
__global__ void k_embed(const int* __restrict__ words, const int* __restrict__ mask,
                        const float* __restrict__ emb, bf16* __restrict__ Xb) {
  int row = blockIdx.x;      // 0..16383
  int k   = threadIdx.x;     // 0..319
  int w   = words[row];
  float mv = (float)mask[row];
  float v = (k < E_) ? emb[(size_t)w * E_ + k] * mv : 0.f;
  Xb[(size_t)row * EP_ + k] = (bf16)v;
}

// ---------------------------------------------------------------- K2: weight conversions
__global__ void k_cvt_wih(const float* __restrict__ Wf, const float* __restrict__ Wb,
                          bf16* __restrict__ dst) {
  int idx = blockIdx.x * blockDim.x + threadIdx.x;  // 2*1024*320
  if (idx >= 2 * G4_ * EP_) return;
  int k = idx % EP_;
  int n = (idx / EP_) % G4_;
  int dir = idx / (EP_ * G4_);
  const float* W = dir ? Wb : Wf;
  float v = (k < E_) ? W[(size_t)n * E_ + k] : 0.f;
  dst[idx] = (bf16)v;
}

__global__ void k_cvt_whh(const float* __restrict__ Wf, const float* __restrict__ Wb,
                          bf16* __restrict__ dst) {
  int idx = blockIdx.x * blockDim.x + threadIdx.x;  // 2*1024*256
  if (idx >= 2 * G4_ * H_) return;
  int dir = idx / (H_ * G4_);
  int rem = idx % (H_ * G4_);
  dst[idx] = (bf16)((dir ? Wb : Wf)[rem]);
}

__global__ void k_bias(const float* bihf, const float* bhhf,
                       const float* bihb, const float* bhhb, float* dst) {
  int idx = blockIdx.x * blockDim.x + threadIdx.x;  // 2048
  if (idx >= 2 * G4_) return;
  int dir = idx / G4_, n = idx % G4_;
  dst[idx] = dir ? (bihb[n] + bhhb[n]) : (bihf[n] + bhhf[n]);
}

// ---------------------------------------------------------------- K3: input projection GEMM (WMMA)
// Gpre[dir][row][n] = sum_k Xb[row][k] * Wih[dir][n][k] + bias[dir][n]
__global__ __launch_bounds__(256) void k_ingemm(const bf16* __restrict__ Xb,
                                                const bf16* __restrict__ Wih,
                                                const float* __restrict__ bias,
                                                float* __restrict__ Gpre) {
  int lane = threadIdx.x & 31;
  int wid  = threadIdx.x >> 5;
  int gw = blockIdx.x * 8 + wid;     // 0..32767 wave-tiles
  int dir = gw >> 14;
  int t   = gw & 16383;
  int tn = t & 31, tm = t >> 5;
  int m0 = tm * 32, n0 = tn * 32;
  const bf16* W = Wih + (size_t)dir * G4_ * EP_;
  const float* bs = bias + dir * G4_;
  float* G = Gpre + (size_t)dir * ROWS_ * G4_;
  int nlane = lane & 15, half = lane >> 4;

  v8f acc[2][2];
  for (int i = 0; i < 2; i++)
    for (int j = 0; j < 2; j++) {
      float bv = bs[n0 + j * 16 + nlane];
      acc[i][j] = (v8f){bv, bv, bv, bv, bv, bv, bv, bv};
    }
  for (int k0 = 0; k0 < EP_; k0 += 32) {
    v16bf a0 = load_frag_a(Xb + (size_t)m0 * EP_ + k0, EP_, lane);
    v16bf a1 = load_frag_a(Xb + (size_t)(m0 + 16) * EP_ + k0, EP_, lane);
    v16bf b0 = load_frag_b(W + (size_t)n0 * EP_ + k0, EP_, lane);
    v16bf b1 = load_frag_b(W + (size_t)(n0 + 16) * EP_ + k0, EP_, lane);
    acc[0][0] = wmma_bf16(a0, b0, acc[0][0]);
    acc[0][1] = wmma_bf16(a0, b1, acc[0][1]);
    acc[1][0] = wmma_bf16(a1, b0, acc[1][0]);
    acc[1][1] = wmma_bf16(a1, b1, acc[1][1]);
  }
  for (int i = 0; i < 2; i++)
    for (int j = 0; j < 2; j++)
      for (int r = 0; r < 8; r++)
        G[(size_t)(m0 + i * 16 + half * 8 + r) * G4_ + n0 + j * 16 + nlane] =
            acc[i][j][r];
}

// ---------------------------------------------------------------- K4: sequential LSTM (one WGP per direction)
// 1024 threads = 32 waves; wave w owns gate columns [32w, 32w+32).
// LDS: gates f32 [64][1024] (256KB) + h bf16 [64][256] (32KB).
__global__ __launch_bounds__(1024) void k_lstm(const bf16* __restrict__ Whh,
                                               const float* __restrict__ Gpre,
                                               float* __restrict__ hout) {
  extern __shared__ char smem[];
  float* gL = (float*)smem;                                  // [64][1024]
  bf16*  hL = (bf16*)(smem + (size_t)B_ * G4_ * 4);          // [64][256]
  int dir = blockIdx.x;
  const bf16* W = Whh + (size_t)dir * G4_ * H_;
  const float* G = Gpre + (size_t)dir * ROWS_ * G4_;
  int tid = threadIdx.x;
  int lane = tid & 31, wid = tid >> 5;
  int nlane = lane & 15, half = lane >> 4;
  int n0 = wid * 32;
  int cb = tid >> 4, cj0 = (tid & 15) * 16;   // cell-state ownership: row cb, cols cj0..cj0+15

  float creg[16];
#pragma unroll
  for (int k = 0; k < 16; k++) creg[k] = 0.f;
  for (int k = 0; k < 16; k++) hL[tid * 16 + k] = (bf16)0.f;
  __syncthreads();

  for (int step = 0; step < S_; step++) {
    int s = dir ? (S_ - 1 - step) : step;
    // init accumulators with precomputed input projection + biases
    v8f acc[4][2];
    for (int i = 0; i < 4; i++)
      for (int j = 0; j < 2; j++) {
        v8f c;
#pragma unroll
        for (int r = 0; r < 8; r++)
          c[r] = G[((size_t)(i * 16 + half * 8 + r) * S_ + s) * G4_ +
                   n0 + j * 16 + nlane];
        acc[i][j] = c;
      }
    // h_{t-1} @ Whh^T via WMMA (A from LDS, B from global/L2)
    for (int k0 = 0; k0 < H_; k0 += 32) {
      v16bf b0 = load_frag_b(W + (size_t)n0 * H_ + k0, H_, lane);
      v16bf b1 = load_frag_b(W + (size_t)(n0 + 16) * H_ + k0, H_, lane);
#pragma unroll
      for (int i = 0; i < 4; i++) {
        v16bf a = load_frag_a(hL + (size_t)(i * 16) * H_ + k0, H_, lane);
        acc[i][0] = wmma_bf16(a, b0, acc[i][0]);
        acc[i][1] = wmma_bf16(a, b1, acc[i][1]);
      }
    }
    // spill gate pre-activations to LDS
    for (int i = 0; i < 4; i++)
      for (int j = 0; j < 2; j++)
#pragma unroll
        for (int r = 0; r < 8; r++)
          gL[(size_t)(i * 16 + half * 8 + r) * G4_ + n0 + j * 16 + nlane] =
              acc[i][j][r];
    __syncthreads();
    // elementwise gate math; gate order i,f,g,o along N
#pragma unroll 4
    for (int k = 0; k < 16; k++) {
      int j = cj0 + k;
      float gi = gL[cb * G4_ + j];
      float gf = gL[cb * G4_ + H_ + j];
      float gg = gL[cb * G4_ + 2 * H_ + j];
      float go = gL[cb * G4_ + 3 * H_ + j];
      float c = sigf(gf) * creg[k] + sigf(gi) * tanhf(gg);
      creg[k] = c;
      float h = sigf(go) * tanhf(c);
      hL[cb * H_ + j] = (bf16)h;
      hout[((size_t)cb * S_ + s) * (2 * H_) + dir * H_ + j] = h;
    }
    __syncthreads();
  }
}

// ---------------------------------------------------------------- K5: tag/tok projection
__global__ void k_proj(const float* __restrict__ hout,
                       const float* __restrict__ Wtag, const float* __restrict__ btag,
                       const float* __restrict__ Wtok, const float* __restrict__ btok,
                       float* __restrict__ tagf, float* __restrict__ tok) {
  int idx = blockIdx.x * blockDim.x + threadIdx.x;  // ROWS_*24
  if (idx >= ROWS_ * 24) return;
  int col = idx % 24, row = idx / 24;
  const float* x = hout + (size_t)row * (2 * H_);
  const float* w;
  float acc;
  if (col < T_) { w = Wtag + (size_t)col * (2 * H_); acc = btag[col]; }
  else          { w = Wtok + (size_t)(col - T_) * (2 * H_); acc = btok[col - T_]; }
  for (int k = 0; k < 2 * H_; k++) acc += x[k] * w[k];
  if (col < T_) tagf[(size_t)row * T_ + col] = acc;
  else          tok[(size_t)row * 2 + (col - T_)] = acc;
}

// ---------------------------------------------------------------- K5b: token cross-entropy
__global__ void k_ce(const float* __restrict__ tok, const int* __restrict__ mask,
                     const int* __restrict__ ltok, float* scal) {
  int idx = blockIdx.x * blockDim.x + threadIdx.x;
  if (idx >= ROWS_) return;
  float a = tok[idx * 2], b = tok[idx * 2 + 1];
  float m = fmaxf(a, b);
  float lse = m + __logf(__expf(a - m) + __expf(b - m));
  int y = ltok[idx];
  float logp = (y ? b : a) - lse;
  float w = (mask[idx] == 1 && y != 0) ? 1.f : 0.f;
  atomicAdd(&scal[0], -logp * w);
  atomicAdd(&scal[1], w);
}

// ---------------------------------------------------------------- K6: CRF forward (logsumexp recursion)
__global__ __launch_bounds__(32) void k_crf_fwd(const float* __restrict__ tagf,
                                                const int* __restrict__ mask,
                                                const float* __restrict__ trans,
                                                float* scal) {
  __shared__ float sh[T_];
  int b = blockIdx.x, j = threadIdx.x;
  float part = -1e30f;
  if (j < T_) part = tagf[(size_t)b * S_ * T_ + j] + trans[STARTT * T_ + j];
  for (int s = 1; s < S_; s++) {
    if (j < T_) sh[j] = part;
    __syncthreads();
    if (j < T_) {
      float m = -1e30f;
      for (int i = 0; i < T_; i++) m = fmaxf(m, sh[i] + trans[i * T_ + j]);
      float sum = 0.f;
      for (int i = 0; i < T_; i++) sum += __expf(sh[i] + trans[i * T_ + j] - m);
      float newp = m + __logf(sum) + tagf[((size_t)b * S_ + s) * T_ + j];
      if (mask[b * S_ + s] > 0) part = newp;
    }
    __syncthreads();
  }
  if (j < T_) sh[j] = part + trans[j * T_ + STOPT];
  __syncthreads();
  if (j == 0) {
    float m = -1e30f;
    for (int i = 0; i < T_; i++) m = fmaxf(m, sh[i]);
    float sum = 0.f;
    for (int i = 0; i < T_; i++) sum += __expf(sh[i] - m);
    atomicAdd(&scal[2], m + __logf(sum));
  }
}

// ---------------------------------------------------------------- K6b: gold path score
__global__ void k_gold(const float* __restrict__ tagf, const int* __restrict__ mask,
                       const int* __restrict__ labels, const float* __restrict__ trans,
                       float* scal) {
  int idx = blockIdx.x * blockDim.x + threadIdx.x;
  if (idx >= ROWS_) return;
  int b = idx / S_, s = idx % S_;
  int tag = labels[idx];
  int prev = (s == 0) ? STARTT : labels[idx - 1];
  float v = (tagf[(size_t)idx * T_ + tag] + trans[prev * T_ + tag]) * (float)mask[idx];
  if (s == 0) {
    int len = 0;
    for (int t = 0; t < S_; t++) len += mask[b * S_ + t];
    int last = labels[b * S_ + (len > 0 ? len - 1 : 0)];
    v += trans[last * T_ + STOPT];
  }
  atomicAdd(&scal[3], v);
}

// ---------------------------------------------------------------- K7: Viterbi + backtrack
__global__ __launch_bounds__(32) void k_viterbi(const float* __restrict__ tagf,
                                                const int* __restrict__ mask,
                                                const float* __restrict__ trans,
                                                int* __restrict__ bp,
                                                float* __restrict__ out_tags) {
  __shared__ float sh[T_];
  int b = blockIdx.x, j = threadIdx.x;
  float vit = -1e30f;
  if (j < T_) vit = tagf[(size_t)b * S_ * T_ + j] + trans[STARTT * T_ + j];
  for (int s = 1; s < S_; s++) {
    if (j < T_) sh[j] = vit;
    __syncthreads();
    if (j < T_) {
      float best = -1e30f;
      int bi = 0;
      for (int i = 0; i < T_; i++) {
        float v = sh[i] + trans[i * T_ + j];
        if (v > best) { best = v; bi = i; }
      }
      int mt = mask[b * S_ + s];
      bp[((size_t)b * (S_ - 1) + (s - 1)) * T_ + j] = (mt > 0) ? bi : j;
      if (mt > 0) vit = best + tagf[((size_t)b * S_ + s) * T_ + j];
    }
    __syncthreads();
  }
  if (j < T_) sh[j] = vit + trans[j * T_ + STOPT];
  __syncthreads();
  if (j == 0) {
    float best = -1e30f;
    int bi = 0;
    for (int i = 0; i < T_; i++)
      if (sh[i] > best) { best = sh[i]; bi = i; }
    int tag = bi;
    out_tags[(size_t)b * S_ + (S_ - 1)] = (float)tag;
    for (int s = S_ - 2; s >= 0; s--) {
      tag = bp[((size_t)b * (S_ - 1) + s) * T_ + tag];
      out_tags[(size_t)b * S_ + s] = (float)tag;
    }
  }
}

// ---------------------------------------------------------------- K8: finalize loss
__global__ void k_final(const float* scal, float* out) {
  if (threadIdx.x == 0 && blockIdx.x == 0) {
    float tok_loss = scal[0] / fmaxf(scal[1], 1.0f);
    out[0] = (scal[2] - scal[3]) / (float)B_ + tok_loss;
  }
}

// ---------------------------------------------------------------- launcher
extern "C" void kernel_launch(void* const* d_in, const int* in_sizes, int n_in,
                              void* d_out, int out_size, void* d_ws, size_t ws_size,
                              hipStream_t stream) {
  const int*   word   = (const int*)d_in[0];
  const int*   mask   = (const int*)d_in[1];
  const int*   labels = (const int*)d_in[2];
  const int*   ltok   = (const int*)d_in[3];
  const float* emb    = (const float*)d_in[4];
  const float* Wih_f  = (const float*)d_in[5];
  const float* Whh_f  = (const float*)d_in[6];
  const float* bih_f  = (const float*)d_in[7];
  const float* bhh_f  = (const float*)d_in[8];
  const float* Wih_b  = (const float*)d_in[9];
  const float* Whh_b  = (const float*)d_in[10];
  const float* bih_b  = (const float*)d_in[11];
  const float* bhh_b  = (const float*)d_in[12];
  const float* Wtag   = (const float*)d_in[13];
  const float* btag   = (const float*)d_in[14];
  const float* Wtok   = (const float*)d_in[15];
  const float* btok   = (const float*)d_in[16];
  const float* trans  = (const float*)d_in[17];

  char* ws = (char*)d_ws;
  size_t off = 0;
  bf16*  Xb    = (bf16*)(ws + off);  off += (size_t)ROWS_ * EP_ * 2;          // 10.5 MB
  bf16*  WihBF = (bf16*)(ws + off);  off += (size_t)2 * G4_ * EP_ * 2;        // 1.3 MB
  bf16*  WhhBF = (bf16*)(ws + off);  off += (size_t)2 * G4_ * H_ * 2;         // 1.0 MB
  float* bias  = (float*)(ws + off); off += (size_t)2 * G4_ * 4;              // 8 KB
  float* Gpre  = (float*)(ws + off); off += (size_t)2 * ROWS_ * G4_ * 4;      // 128 MB
  float* hout  = (float*)(ws + off); off += (size_t)ROWS_ * 2 * H_ * 4;       // 32 MB
  float* tagf  = (float*)(ws + off); off += (size_t)ROWS_ * T_ * 4;           // 1.4 MB
  float* tok   = (float*)(ws + off); off += (size_t)ROWS_ * 2 * 4;            // 128 KB
  int*   bp    = (int*)(ws + off);   off += (size_t)B_ * (S_ - 1) * T_ * 4;   // 1.4 MB
  float* scal  = (float*)(ws + off); off += 4 * 4;

  float* out = (float*)d_out;

  k_zero<<<1, 32, 0, stream>>>(scal);
  k_embed<<<ROWS_, EP_, 0, stream>>>(word, mask, emb, Xb);
  k_cvt_wih<<<(2 * G4_ * EP_ + 255) / 256, 256, 0, stream>>>(Wih_f, Wih_b, WihBF);
  k_cvt_whh<<<(2 * G4_ * H_ + 255) / 256, 256, 0, stream>>>(Whh_f, Whh_b, WhhBF);
  k_bias<<<(2 * G4_ + 255) / 256, 256, 0, stream>>>(bih_f, bhh_f, bih_b, bhh_b, bias);

  // input-projection GEMM: 2 dirs * (16384/32)*(1024/32) = 32768 wave-tiles, 8 waves/block
  k_ingemm<<<4096, 256, 0, stream>>>(Xb, WihBF, bias, Gpre);

  // recurrent LSTM: one 32-wave workgroup per direction; 288 KB dynamic LDS
  size_t lds = (size_t)B_ * G4_ * 4 + (size_t)B_ * H_ * 2;
  k_lstm<<<2, 1024, lds, stream>>>(WhhBF, Gpre, hout);

  k_proj<<<(ROWS_ * 24 + 255) / 256, 256, 0, stream>>>(hout, Wtag, btag, Wtok, btok,
                                                       tagf, tok);
  k_ce<<<(ROWS_ + 255) / 256, 256, 0, stream>>>(tok, mask, ltok, scal);
  k_crf_fwd<<<B_, 32, 0, stream>>>(tagf, mask, trans, scal);
  k_gold<<<(ROWS_ + 255) / 256, 256, 0, stream>>>(tagf, mask, labels, trans, scal);
  k_viterbi<<<B_, 32, 0, stream>>>(tagf, mask, trans, bp, out + 1);
  k_final<<<1, 1, 0, stream>>>(scal, out);
}
